// MoSRNet_32100585571065
// MI455X (gfx1250) — matrix operand order repaired and checked by reading, not compiled, not run on hardware
//
#include <hip/hip_runtime.h>
#include <hip/hip_bf16.h>

// ---------------------------------------------------------------------------
// MoSRNet: S=3 independent subnets on B=32768 rows.
//   conv1d(1->32,k3,pad1) -> gelu -> conv1d(32->64,k3,pad1) -> gelu
//   -> BatchNorm (training stats over B*L) -> flatten(320) -> Linear(320->541)
// BN is folded into the final Linear; the big GEMM runs on bf16 WMMA.
// ---------------------------------------------------------------------------

#define Bsz   32768
#define Ssz   3
#define Lsz   5
#define D1sz  32
#define D2sz  64
#define OUTsz 541
#define Kdim  (D2sz * Lsz)        // 320
#define NPAD  576                 // 541 padded to 9 * 64
#define BBLK  64                  // b-rows per feat block
#define NBBLK (Bsz / BBLK)        // 512 feat blocks per subnet

typedef __attribute__((ext_vector_type(16))) __bf16 v16bf;
typedef __attribute__((ext_vector_type(8)))  float  v8f;

struct alignas(16) U128 { unsigned a, b, c, d; };
union BFrag { U128 u[2]; v16bf v; };

__device__ __forceinline__ float gelu_exact(float x) {
    return 0.5f * x * (1.0f + erff(x * 0.70710678118654752f));
}

__device__ __forceinline__ unsigned short f32_to_bf16(float f) {
    unsigned u = __float_as_uint(f);
    unsigned r = u + 0x7FFFu + ((u >> 16) & 1u);   // round-to-nearest-even
    return (unsigned short)(r >> 16);
}

// ---------------------------------------------------------------------------
// Kernel 1: conv1 -> gelu -> conv2 -> gelu, emit bf16 activations + stats
// grid = (512, 3), block = 256
// ---------------------------------------------------------------------------
__global__ void __launch_bounds__(256)
feat_kernel(const float* __restrict__ x,  const float* __restrict__ w1,
            const float* __restrict__ b1, const float* __restrict__ w2,
            const float* __restrict__ b2,
            unsigned short* __restrict__ Abf, float* __restrict__ part) {
    __shared__ float xs[BBLK * Lsz];          // 1.25 KB
    __shared__ float h1[BBLK * D1sz * Lsz];   // 40 KB
    __shared__ float w1s[D1sz * 3];
    __shared__ float b1s[D1sz];
    __shared__ float b2s[D2sz];
    __shared__ float red[256 * 2];

    const int s  = blockIdx.y;
    const int bx = blockIdx.x;
    const long b0 = (long)bx * BBLK;
    const int t  = threadIdx.x;

    for (int e = t; e < BBLK * Lsz; e += 256) {
        int i = e / Lsz, l = e % Lsz;
        xs[e] = x[(b0 + i) * (Ssz * Lsz) + s * Lsz + l];
    }
    if (t < D1sz * 3) w1s[t] = w1[s * D1sz * 3 + t];
    if (t < D1sz)     b1s[t] = b1[s * D1sz + t];
    if (t < D2sz)     b2s[t] = b2[s * D2sz + t];
    __syncthreads();

    // ---- phase A: h1[b, c1, l] = gelu(conv1) ----
    for (int it = 0; it < (BBLK * D1sz) / 256; ++it) {
        int task = it * 256 + t;
        int c1 = task & (D1sz - 1);
        int i  = task >> 5;
        const float* xi = &xs[i * Lsz];
        float k0 = w1s[c1 * 3 + 0], k1 = w1s[c1 * 3 + 1], k2 = w1s[c1 * 3 + 2];
        float bb = b1s[c1];
        float* hp = &h1[i * (D1sz * Lsz) + c1 * Lsz];
#pragma unroll
        for (int l = 0; l < Lsz; ++l) {
            float v = bb + k1 * xi[l];
            if (l > 0)       v += k0 * xi[l - 1];
            if (l < Lsz - 1) v += k2 * xi[l + 1];
            hp[l] = gelu_exact(v);
        }
    }
    __syncthreads();

    // ---- phase B: h2 = gelu(conv2), write bf16, accumulate stats ----
    float s1 = 0.f, s2 = 0.f;
    const float* w2base = w2 + (size_t)s * D2sz * D1sz * 3;
    for (int it = 0; it < (BBLK * D2sz) / 256; ++it) {
        int task = it * 256 + t;
        int c2 = task & (D2sz - 1);           // constant per thread
        int i  = task >> 6;
        const float* hb = &h1[i * (D1sz * Lsz)];
        const float* wc = w2base + c2 * (D1sz * 3);
        float acc[Lsz];
#pragma unroll
        for (int l = 0; l < Lsz; ++l) acc[l] = b2s[c2];
        for (int c1 = 0; c1 < D1sz; ++c1) {
            float k0 = wc[c1 * 3 + 0], k1 = wc[c1 * 3 + 1], k2 = wc[c1 * 3 + 2];
            const float* hh = &hb[c1 * Lsz];
#pragma unroll
            for (int l = 0; l < Lsz; ++l) {
                float v = k1 * hh[l];
                if (l > 0)       v += k0 * hh[l - 1];
                if (l < Lsz - 1) v += k2 * hh[l + 1];
                acc[l] += v;
            }
        }
        unsigned short* ap =
            Abf + ((size_t)s * Bsz + b0 + i) * Kdim + c2 * Lsz;
#pragma unroll
        for (int l = 0; l < Lsz; ++l) {
            float g = gelu_exact(acc[l]);
            ap[l] = f32_to_bf16(g);
            s1 += g;
            s2 += g * g;
        }
    }
    red[t * 2]     = s1;
    red[t * 2 + 1] = s2;
    __syncthreads();
    if (t < D2sz) {   // deterministic per-block partial (channel = t)
        float a1 = red[t * 2]         + red[(t + 64) * 2] +
                   red[(t + 128) * 2] + red[(t + 192) * 2];
        float a2 = red[t * 2 + 1]         + red[(t + 64) * 2 + 1] +
                   red[(t + 128) * 2 + 1] + red[(t + 192) * 2 + 1];
        float* pp = part + ((size_t)(s * NBBLK + bx) * D2sz + t) * 2;
        pp[0] = a1;
        pp[1] = a2;
    }
}

// ---------------------------------------------------------------------------
// Kernel 2: deterministic reduction of per-block partials -> stats[s][c][2]
// grid = 3, block = 64
// ---------------------------------------------------------------------------
__global__ void __launch_bounds__(64)
reduce_kernel(const float* __restrict__ part, float* __restrict__ stats) {
    int s = blockIdx.x, t = threadIdx.x;
    float a1 = 0.f, a2 = 0.f;
    for (int j = 0; j < NBBLK; ++j) {
        const float* pp = part + ((size_t)(s * NBBLK + j) * D2sz + t) * 2;
        a1 += pp[0];
        a2 += pp[1];
    }
    stats[(s * D2sz + t) * 2]     = a1;
    stats[(s * D2sz + t) * 2 + 1] = a2;
}

// ---------------------------------------------------------------------------
// Kernel 3: fold BN into Linear: W'[n,k]=wl[n,k]*scale[k/5] (bf16, N padded),
//           b'[n]=bl[n]+sum_k shift[k/5]*wl[n,k]
// grid = 3*576, block = 64 (thread t == channel c, k in [5t,5t+5))
// ---------------------------------------------------------------------------
__global__ void __launch_bounds__(64)
fold_kernel(const float* __restrict__ wl, const float* __restrict__ bl,
            const float* __restrict__ gamma, const float* __restrict__ beta,
            const float* __restrict__ stats,
            unsigned short* __restrict__ Wbf, float* __restrict__ blp) {
    __shared__ float red[64];
    const int s = blockIdx.x / NPAD;
    const int n = blockIdx.x % NPAD;
    const int t = threadIdx.x;

    const float inv   = 1.0f / (float)(Bsz * Lsz);
    const float mean  = stats[(s * D2sz + t) * 2] * inv;
    const float var   = stats[(s * D2sz + t) * 2 + 1] * inv - mean * mean;
    const float scale = gamma[s * D2sz + t] * rsqrtf(var + 1e-5f);
    const float shift = beta[s * D2sz + t] - mean * scale;

    unsigned short* wp = Wbf + ((size_t)s * NPAD + n) * Kdim + t * Lsz;
    float partial = 0.f;
    if (n < OUTsz) {
        const float* wlp = wl + ((size_t)s * OUTsz + n) * Kdim + t * Lsz;
        float sum = 0.f;
#pragma unroll
        for (int l = 0; l < Lsz; ++l) {
            float w = wlp[l];
            wp[l] = f32_to_bf16(w * scale);
            sum += w;
        }
        partial = shift * sum;
    } else {
#pragma unroll
        for (int l = 0; l < Lsz; ++l) wp[l] = 0;
    }
    red[t] = partial;
    __syncthreads();
    for (int off = 32; off > 0; off >>= 1) {
        if (t < off) red[t] += red[t + off];
        __syncthreads();
    }
    if (t == 0)
        blp[s * NPAD + n] = (n < OUTsz ? bl[s * OUTsz + n] : 0.f) + red[0];
}

// ---------------------------------------------------------------------------
// Kernel 4: GEMM  out[b, s, n] = A[s][b, :320] . W'[s][n, :320] + b'[s][n]
// wave tile 32(M) x 64(N): 2 A-frags x 4 B-frags, 8 WMMAs / K-step, 10 steps.
// grid = 3456 blocks x 256 threads (8 independent waves per block).
// ---------------------------------------------------------------------------
__global__ void __launch_bounds__(256)
gemm_kernel(const unsigned short* __restrict__ Abf,
            const unsigned short* __restrict__ Wbf,
            const float* __restrict__ blp, float* __restrict__ out) {
    const int lane = threadIdx.x & 31;
    const int wave = threadIdx.x >> 5;
    const int tile = blockIdx.x * 8 + wave;      // 27648 total, exact

    const int MT = Bsz / 32;                     // 1024
    const int NT = NPAD / 64;                    // 9
    const int s   = tile / (MT * NT);
    const int rem = tile % (MT * NT);
    const long m0 = (long)(rem / NT) * 32;
    const int  n0 = (rem % NT) * 64;

    const int lr = lane & 15;
    const int hi = lane >> 4;

    v8f acc[2][4] = {};

    const unsigned short* Abase0 =
        Abf + ((size_t)s * Bsz + m0 + lr) * Kdim;
    const unsigned short* Abase1 = Abase0 + (size_t)16 * Kdim;
    const unsigned short* Bbase[4];
#pragma unroll
    for (int ni = 0; ni < 4; ++ni)
        Bbase[ni] = Wbf + ((size_t)s * NPAD + n0 + ni * 16 + lr) * Kdim;

    for (int kk = 0; kk < Kdim; kk += 32) {
        BFrag a0, a1, bfr[4];
        // A 16x32 bf16: lane<16 holds K {0..7,16..23}, lane>=16 {8..15,24..31}
        const unsigned short* pa0 = Abase0 + kk + hi * 8;
        const unsigned short* pa1 = Abase1 + kk + hi * 8;
        a0.u[0] = *(const U128*)pa0;
        a0.u[1] = *(const U128*)(pa0 + 16);
        a1.u[0] = *(const U128*)pa1;
        a1.u[1] = *(const U128*)(pa1 + 16);
        // B 32x16 bf16: lane<16 holds K {0..7,16..23} of column N, lane>=16 {8..15,24..31}
#pragma unroll
        for (int ni = 0; ni < 4; ++ni) {
            const unsigned short* pb = Bbase[ni] + kk + hi * 8;
            bfr[ni].u[0] = *(const U128*)pb;
            bfr[ni].u[1] = *(const U128*)(pb + 16);
        }
#pragma unroll
        for (int ni = 0; ni < 4; ++ni) {
            acc[0][ni] = __builtin_amdgcn_wmma_f32_16x16x32_bf16(
                false, a0.v, false, bfr[ni].v, (short)0, acc[0][ni],
                false, false);
            acc[1][ni] = __builtin_amdgcn_wmma_f32_16x16x32_bf16(
                false, a1.v, false, bfr[ni].v, (short)0, acc[1][ni],
                false, false);
        }
    }

    // epilogue: + folded bias, store (guard n < 541)
#pragma unroll
    for (int ni = 0; ni < 4; ++ni) {
        const int col = n0 + ni * 16 + lr;
        if (col >= OUTsz) continue;
        const float bias = blp[s * NPAD + col];
#pragma unroll
        for (int mi = 0; mi < 2; ++mi) {
#pragma unroll
            for (int v = 0; v < 8; ++v) {
                size_t row = (size_t)m0 + mi * 16 + hi * 8 + v;
                out[row * (Ssz * OUTsz) + (size_t)s * OUTsz + col] =
                    acc[mi][ni][v] + bias;
            }
        }
    }
}

// ---------------------------------------------------------------------------
// Workspace layout (bytes):
//   stats : [3][64][2] f32                  @ 0        (1.5 KB)
//   blp   : [3][576]  f32                   @ 0x1000
//   part  : [3][512][64][2] f32             @ 0x2000   (768 KB)
//   Wbf   : [3][576][320] bf16              @ 0x100000 (1.06 MB)
//   Abf   : [3][32768][320] bf16            @ 0x210000 (60 MB)
// ---------------------------------------------------------------------------
extern "C" void kernel_launch(void* const* d_in, const int* in_sizes, int n_in,
                              void* d_out, int out_size, void* d_ws,
                              size_t ws_size, hipStream_t stream) {
    const float* x     = (const float*)d_in[0];
    const float* w1    = (const float*)d_in[1];
    const float* b1    = (const float*)d_in[2];
    const float* w2    = (const float*)d_in[3];
    const float* b2    = (const float*)d_in[4];
    const float* gamma = (const float*)d_in[5];
    const float* beta  = (const float*)d_in[6];
    const float* wl    = (const float*)d_in[7];
    const float* bl    = (const float*)d_in[8];
    float* out = (float*)d_out;

    char* ws = (char*)d_ws;
    float*          stats = (float*)(ws + 0x0);
    float*          blp   = (float*)(ws + 0x1000);
    float*          part  = (float*)(ws + 0x2000);
    unsigned short* Wbf   = (unsigned short*)(ws + 0x100000);
    unsigned short* Abf   = (unsigned short*)(ws + 0x210000);

    feat_kernel<<<dim3(NBBLK, Ssz), 256, 0, stream>>>(x, w1, b1, w2, b2,
                                                      Abf, part);
    reduce_kernel<<<Ssz, 64, 0, stream>>>(part, stats);
    fold_kernel<<<Ssz * NPAD, 64, 0, stream>>>(wl, bl, gamma, beta, stats,
                                               Wbf, blp);
    gemm_kernel<<<(Bsz / 32) * (NPAD / 64) * Ssz / 8, 256, 0, stream>>>(
        Abf, Wbf, blp, out);
}
